// FullAttention_27814208209698
// MI455X (gfx1250) — compile-verified, matrix operand order checked by reading, not run on hardware
//
#include <hip/hip_runtime.h>
#include <hip/hip_bf16.h>
#include <math.h>

// ---------- problem constants ----------
#define DM   1024
#define NH   16
#define HD   64
#define BATCH 4
#define SEQ  2048
#define ROWS (BATCH * SEQ)          // 8192 token rows

// ---------- vector types ----------
typedef __attribute__((ext_vector_type(16))) __bf16 v16bf;
typedef __attribute__((ext_vector_type(8)))  float  v8f;
typedef __attribute__((ext_vector_type(4)))  unsigned u4v;
typedef __attribute__((ext_vector_type(2)))  unsigned u2v;
typedef __attribute__((ext_vector_type(4)))  float    f4v;

union BFrag { v16bf v; u4v u[2]; unsigned wd[8]; };

// fp32 -> bf16 round-to-nearest-even
__device__ __forceinline__ unsigned short f2bf(float f) {
  unsigned u = __float_as_uint(f);
  u += 0x7FFFu + ((u >> 16) & 1u);
  return (unsigned short)(u >> 16);
}
__device__ __forceinline__ unsigned pk2(unsigned short a, unsigned short b) {
  return (unsigned)a | ((unsigned)b << 16);
}
__device__ __forceinline__ float fast_exp2(float x) {
  return __builtin_amdgcn_exp2f(x);   // v_exp_f32
}
__device__ __forceinline__ v8f wmma_bf16(v16bf a, v16bf b, v8f c) {
  return __builtin_amdgcn_wmma_f32_16x16x32_bf16(false, a, false, b, (short)0, c,
                                                 false, false);
}

// ---------------------------------------------------------------------------
// Kernel 1: W fp32 [K=1024, N=1024] row-major  ->  Wt bf16 [N, K] row-major
// ---------------------------------------------------------------------------
__global__ void wtrans_kernel(const float* __restrict__ W,
                              unsigned short* __restrict__ WT) {
  __shared__ float tile[32][33];
  const int n0 = blockIdx.x * 32;
  const int k0 = blockIdx.y * 32;
  const int tx = threadIdx.x;     // 0..31
  const int ty = threadIdx.y;     // 0..7
#pragma unroll
  for (int i = 0; i < 4; ++i) {
    int k = k0 + ty + i * 8;
    tile[ty + i * 8][tx] = W[(size_t)k * DM + n0 + tx];
  }
  __syncthreads();
#pragma unroll
  for (int i = 0; i < 4; ++i) {
    int n = n0 + ty + i * 8;
    WT[(size_t)n * DM + k0 + tx] = f2bf(tile[tx][ty + i * 8]);
  }
}

// ---------------------------------------------------------------------------
// Kernel 2: tiled WMMA GEMM  C[i,j] = A[i,:] * Wt[j,:]^T + bias[j]
//   block = 256 threads (8 waves), tile 128x128, BK = 32
//   MODE 0: store bf16 to [b,h,l,d]   (Q / K; outScale folded in)
//   MODE 1: store bf16 to [b,h,d,s]   (V transposed)
//   MODE 2: store fp32 to [i,j]       (final output)
// ---------------------------------------------------------------------------
template <bool A_BF16, int MODE>
__global__ void __launch_bounds__(256)
gemm_kernel(const void* __restrict__ Ain, const unsigned short* __restrict__ Bt,
            const float* __restrict__ bias, void* __restrict__ Out,
            float outScale) {
  __shared__ unsigned short As[128][40];   // pad: 80B row stride (16B aligned)
  __shared__ unsigned short Bs[128][40];

  const int t    = threadIdx.x;
  const int lane = t & 31;
  const int ln   = lane & 15;
  const int hi8  = (lane >> 4) << 3;       // 0 or 8 (K / M sub-offset)
  const int w    = t >> 5;
  const int wr   = w & 1;
  const int wc   = w >> 1;
  const int i0   = blockIdx.y * 128;
  const int j0   = blockIdx.x * 128;

  v8f acc[8];
#pragma unroll
  for (int a = 0; a < 8; ++a) acc[a] = (v8f){0, 0, 0, 0, 0, 0, 0, 0};

  for (int kt = 0; kt < DM; kt += 32) {
    // ---- stage A tile (128 x 32) into LDS as bf16 ----
    if (A_BF16) {
      const unsigned short* Ag = (const unsigned short*)Ain;
#pragma unroll
      for (int it = 0; it < 2; ++it) {
        int r = (t >> 2) + it * 64;
        int ch = t & 3;
        u4v x = *(const u4v*)(Ag + (size_t)(i0 + r) * DM + kt + ch * 8);
        *(u4v*)&As[r][ch * 8] = x;
      }
      if (kt + 32 < DM)
        __builtin_prefetch(Ag + (size_t)(i0 + (t >> 2)) * DM + kt + 32, 0, 3);
    } else {
      const float* Ag = (const float*)Ain;
#pragma unroll
      for (int it = 0; it < 4; ++it) {
        int r = (t >> 3) + it * 32;
        int c4 = t & 7;
        f4v x = *(const f4v*)(Ag + (size_t)(i0 + r) * DM + kt + c4 * 4);
        u2v p;
        p.x = pk2(f2bf(x.x), f2bf(x.y));
        p.y = pk2(f2bf(x.z), f2bf(x.w));
        *(u2v*)&As[r][c4 * 4] = p;
      }
      if (kt + 32 < DM)
        __builtin_prefetch(Ag + (size_t)(i0 + (t >> 3)) * DM + kt + 32, 0, 3);
    }
    // ---- stage Wt tile (128 x 32) ----
    {
#pragma unroll
      for (int it = 0; it < 2; ++it) {
        int r = (t >> 2) + it * 64;
        int ch = t & 3;
        u4v x = *(const u4v*)(Bt + (size_t)(j0 + r) * DM + kt + ch * 8);
        *(u4v*)&Bs[r][ch * 8] = x;
      }
      if (kt + 32 < DM)
        __builtin_prefetch(Bt + (size_t)(j0 + (t >> 2)) * DM + kt + 32, 0, 3);
    }
    __syncthreads();

    // ---- fragments + 8 WMMAs ----
    BFrag aF[4], bF[2];
#pragma unroll
    for (int mt = 0; mt < 4; ++mt) {
      const unsigned short* p = &As[wr * 64 + mt * 16 + ln][0];
      aF[mt].u[0] = *(const u4v*)(p + hi8);
      aF[mt].u[1] = *(const u4v*)(p + 16 + hi8);
    }
#pragma unroll
    for (int nt = 0; nt < 2; ++nt) {
      const unsigned short* p = &Bs[wc * 32 + nt * 16 + ln][0];
      bF[nt].u[0] = *(const u4v*)(p + hi8);
      bF[nt].u[1] = *(const u4v*)(p + 16 + hi8);
    }
#pragma unroll
    for (int mt = 0; mt < 4; ++mt)
#pragma unroll
      for (int nt = 0; nt < 2; ++nt)
        acc[mt * 2 + nt] = wmma_bf16(aF[mt].v, bF[nt].v, acc[mt * 2 + nt]);
    __syncthreads();
  }

  // ---- epilogue: lane holds 8 consecutive rows (M) of one column (N) ----
#pragma unroll
  for (int mt = 0; mt < 4; ++mt) {
#pragma unroll
    for (int nt = 0; nt < 2; ++nt) {
      v8f c = acc[mt * 2 + nt];
      const int ibase = i0 + wr * 64 + mt * 16 + hi8;   // first of 8 rows
      const int j = j0 + wc * 32 + nt * 16 + ln;
      const float bv = bias[j];
      if (MODE == 0) {
        unsigned short* O = (unsigned short*)Out;
        const int h = j >> 6, d = j & 63;
#pragma unroll
        for (int e = 0; e < 8; ++e) {
          int i = ibase + e;
          int b = i >> 11, l = i & 2047;
          size_t addr = ((size_t)((b * NH + h) * SEQ + l)) * HD + d;
          O[addr] = f2bf((c[e] + bv) * outScale);
        }
      } else if (MODE == 1) {
        unsigned short* O = (unsigned short*)Out;
        const int h = j >> 6, d = j & 63;
        const int b = ibase >> 11, l = ibase & 2047;   // 8 rows share b
        size_t base = ((size_t)((b * NH + h) * HD + d)) * SEQ + l;
        u4v pkt;
        pkt.x = pk2(f2bf(c[0] + bv), f2bf(c[1] + bv));
        pkt.y = pk2(f2bf(c[2] + bv), f2bf(c[3] + bv));
        pkt.z = pk2(f2bf(c[4] + bv), f2bf(c[5] + bv));
        pkt.w = pk2(f2bf(c[6] + bv), f2bf(c[7] + bv));
        *(u4v*)(O + base) = pkt;
      } else {
        float* O = (float*)Out;
#pragma unroll
        for (int e = 0; e < 8; ++e)
          O[(size_t)(ibase + e) * DM + j] = c[e] + bv;
      }
    }
  }
}

// ---------------------------------------------------------------------------
// Kernel 3: flash attention, all-register per-wave 16-row Q block.
//   scoresT = K . Q^T  (A = K rows, B = Q rows) -> D-layout of scoresT
//   matches the B-layout required for  outT = V^T . P  directly.
//   Q was pre-scaled by SCALE*log2(e) -> exp2 softmax.
// ---------------------------------------------------------------------------
__global__ void __launch_bounds__(128)
flash_kernel(const unsigned short* __restrict__ Q,
             const unsigned short* __restrict__ K,
             const unsigned short* __restrict__ VT,
             unsigned short* __restrict__ ATTN) {
  const int t    = threadIdx.x;
  const int lane = t & 31;
  const int ln   = lane & 15;
  const int hi8  = (lane >> 4) << 3;
  const int w    = t >> 5;
  const int bh   = blockIdx.y;
  const int b    = bh >> 4, h = bh & 15;
  const int l0   = blockIdx.x * 64 + w * 16;

  // Q fragments (B operand, lane ln = column l): row-major b128 loads
  BFrag qf[2];
  {
    const unsigned short* qrow = Q + ((size_t)bh * SEQ + l0 + ln) * HD;
    qf[0].u[0] = *(const u4v*)(qrow + hi8);
    qf[0].u[1] = *(const u4v*)(qrow + 16 + hi8);
    qf[1].u[0] = *(const u4v*)(qrow + 32 + hi8);
    qf[1].u[1] = *(const u4v*)(qrow + 48 + hi8);
  }

  v8f acc[4];
#pragma unroll
  for (int a = 0; a < 4; ++a) acc[a] = (v8f){0, 0, 0, 0, 0, 0, 0, 0};
  float m = -INFINITY, rsum = 0.0f;
  const v8f zero = (v8f){0, 0, 0, 0, 0, 0, 0, 0};

  for (int s0 = 0; s0 < SEQ; s0 += 32) {
    // K fragments: two 16-row sub-tiles x two 32-wide k(d) steps
    const unsigned short* kr0 = K + ((size_t)bh * SEQ + s0 + ln) * HD;
    const unsigned short* kr1 = kr0 + 16 * HD;
    BFrag ka0, ka1, kb0, kb1;
    ka0.u[0] = *(const u4v*)(kr0 + hi8);      ka0.u[1] = *(const u4v*)(kr0 + 16 + hi8);
    ka1.u[0] = *(const u4v*)(kr0 + 32 + hi8); ka1.u[1] = *(const u4v*)(kr0 + 48 + hi8);
    kb0.u[0] = *(const u4v*)(kr1 + hi8);      kb0.u[1] = *(const u4v*)(kr1 + 16 + hi8);
    kb1.u[0] = *(const u4v*)(kr1 + 32 + hi8); kb1.u[1] = *(const u4v*)(kr1 + 48 + hi8);

    v8f d1 = wmma_bf16(ka0.v, qf[0].v, zero);
    d1     = wmma_bf16(ka1.v, qf[1].v, d1);
    v8f d2 = wmma_bf16(kb0.v, qf[0].v, zero);
    d2     = wmma_bf16(kb1.v, qf[1].v, d2);

    // online softmax over this 32-column chunk (per output column l)
    float mx = d1[0];
#pragma unroll
    for (int e = 1; e < 8; ++e) mx = fmaxf(mx, d1[e]);
#pragma unroll
    for (int e = 0; e < 8; ++e) mx = fmaxf(mx, d2[e]);
    mx = fmaxf(mx, __shfl_xor(mx, 16, 32));
    const float mnew  = fmaxf(m, mx);
    const float alpha = fast_exp2(m - mnew);

    float pe[16];
    float s = 0.0f;
#pragma unroll
    for (int e = 0; e < 8; ++e) { pe[e] = fast_exp2(d1[e] - mnew); s += pe[e]; }
#pragma unroll
    for (int e = 0; e < 8; ++e) { pe[8 + e] = fast_exp2(d2[e] - mnew); s += pe[8 + e]; }
    s += __shfl_xor(s, 16, 32);
    rsum = rsum * alpha + s;
    m = mnew;

    BFrag pf;   // D-layout of scoresT == B-layout for the PV WMMA
#pragma unroll
    for (int i = 0; i < 8; ++i) pf.wd[i] = pk2(f2bf(pe[2 * i]), f2bf(pe[2 * i + 1]));

#pragma unroll
    for (int a = 0; a < 4; ++a) acc[a] *= alpha;

    // outT[d, l] += V^T[d, s-chunk] . P[s-chunk, l]
#pragma unroll
    for (int db = 0; db < 4; ++db) {
      const unsigned short* vr =
          VT + ((size_t)bh * HD + db * 16 + ln) * SEQ + s0;
      BFrag vf;
      vf.u[0] = *(const u4v*)(vr + hi8);
      vf.u[1] = *(const u4v*)(vr + 16 + hi8);
      acc[db] = wmma_bf16(vf.v, pf.v, acc[db]);
    }
  }

  const float inv = 1.0f / rsum;
#pragma unroll
  for (int db = 0; db < 4; ++db) {
    v8f c = acc[db] * inv;
    u4v pkt;
    pkt.x = pk2(f2bf(c[0]), f2bf(c[1]));
    pkt.y = pk2(f2bf(c[2]), f2bf(c[3]));
    pkt.z = pk2(f2bf(c[4]), f2bf(c[5]));
    pkt.w = pk2(f2bf(c[6]), f2bf(c[7]));
    size_t base =
        ((size_t)(b * SEQ + l0 + ln)) * DM + h * HD + db * 16 + hi8;
    *(u4v*)(ATTN + base) = pkt;
  }
}

// ---------------------------------------------------------------------------
extern "C" void kernel_launch(void* const* d_in, const int* in_sizes, int n_in,
                              void* d_out, int out_size, void* d_ws,
                              size_t ws_size, hipStream_t stream) {
  (void)in_sizes; (void)n_in; (void)out_size; (void)ws_size;
  const float* Xq = (const float*)d_in[0];
  const float* Xk = (const float*)d_in[1];
  const float* Xv = (const float*)d_in[2];
  const float* Wq = (const float*)d_in[3];
  const float* bq = (const float*)d_in[4];
  const float* Wk = (const float*)d_in[5];
  const float* bk = (const float*)d_in[6];
  const float* Wv = (const float*)d_in[7];
  const float* bv = (const float*)d_in[8];
  const float* Wo = (const float*)d_in[9];
  const float* bo = (const float*)d_in[10];
  float* out = (float*)d_out;

  // workspace layout (bf16 halves): 4 x 1M weights + 4 x 8M tensors = 72 MB
  unsigned short* WqT = (unsigned short*)d_ws;
  unsigned short* WkT = WqT + (size_t)DM * DM;
  unsigned short* WvT = WkT + (size_t)DM * DM;
  unsigned short* WoT = WvT + (size_t)DM * DM;
  unsigned short* Qb  = WoT + (size_t)DM * DM;
  unsigned short* Kb  = Qb + (size_t)ROWS * DM;
  unsigned short* Vt  = Kb + (size_t)ROWS * DM;
  unsigned short* At  = Vt + (size_t)ROWS * DM;

  dim3 ttb(32, 8), ttg(DM / 32, DM / 32);
  wtrans_kernel<<<ttg, ttb, 0, stream>>>(Wq, WqT);
  wtrans_kernel<<<ttg, ttb, 0, stream>>>(Wk, WkT);
  wtrans_kernel<<<ttg, ttb, 0, stream>>>(Wv, WvT);
  wtrans_kernel<<<ttg, ttb, 0, stream>>>(Wo, WoT);

  dim3 gg(DM / 128, ROWS / 128);  // (8, 64)
  const float qscale = 0.125f * 1.4426950408889634f;  // HD^-0.5 * log2(e)
  gemm_kernel<false, 0><<<gg, 256, 0, stream>>>(Xq, WqT, bq, Qb, qscale);
  gemm_kernel<false, 0><<<gg, 256, 0, stream>>>(Xk, WkT, bk, Kb, 1.0f);
  gemm_kernel<false, 1><<<gg, 256, 0, stream>>>(Xv, WvT, bv, Vt, 1.0f);

  flash_kernel<<<dim3(SEQ / 64, BATCH * NH), 128, 0, stream>>>(Qb, Kb, Vt, At);

  gemm_kernel<true, 2><<<gg, 256, 0, stream>>>(At, WoT, bo, out, 1.0f);
}
